// GNNWithAttentionFusion_45621142618696
// MI455X (gfx1250) — compile-verified, matrix operand order checked by reading, not run on hardware
//
#include <hip/hip_runtime.h>
#include <math.h>

// ---------------- problem constants (match reference) ----------------
#define NNODES 131072
#define NEDGES 1048576
#define ETOT   (NEDGES + NNODES)   // edges + self loops
#define INC    256
#define HC     128
#define OUTC   64
#define NG     128
#define NPG    1024

typedef __attribute__((ext_vector_type(16))) _Float16 v16h;
typedef __attribute__((ext_vector_type(8)))  float    v8f;
typedef __attribute__((ext_vector_type(4)))  int      v4i;

union HFrag { v16h v; unsigned u[8]; };
union H4    { _Float16 h[4]; uint2 u2; };

enum { ACT_NONE = 0, ACT_RELU = 1, ACT_SIGMOID = 2 };

// ---------- CDNA5 async global->LDS copy (ASYNCcnt-tracked) ----------
#if defined(__has_builtin)
#  if __has_builtin(__builtin_amdgcn_global_load_async_to_lds_b128)
#    define USE_ASYNC_LDS 1
#  endif
#  if __has_builtin(__builtin_amdgcn_s_wait_asynccnt)
#    define HAVE_WAIT_ASYNC 1
#  endif
#endif
#ifndef USE_ASYNC_LDS
#define USE_ASYNC_LDS 0
#endif
#ifndef HAVE_WAIT_ASYNC
#define HAVE_WAIT_ASYNC 0
#endif

typedef __attribute__((address_space(1))) v4i* gptr_v4i;   // global int4*
typedef __attribute__((address_space(3))) v4i* lptr_v4i;   // LDS int4*

__device__ __forceinline__ void copy16_g2lds(const void* g, void* l)
{
#if USE_ASYNC_LDS
    __builtin_amdgcn_global_load_async_to_lds_b128(
        (gptr_v4i)(unsigned long long)g,
        (lptr_v4i)(unsigned int)(unsigned long long)l,
        0, 0);
#else
    *(uint4*)l = *(const uint4*)g;
#endif
}

__device__ __forceinline__ void wait_async_copies()
{
#if USE_ASYNC_LDS
#  if HAVE_WAIT_ASYNC
    __builtin_amdgcn_s_wait_asynccnt(0);
#  else
    asm volatile("s_wait_asynccnt 0" ::: "memory");
#  endif
#endif
}

// ------------------------------------------------------------------
// WMMA GEMM:  C[M,NCOLS] = act(A[M,K] @ Bt^T + bias)
//   A : f32 (AF16=0) or f16 (AF16=1), row stride lda (elements)
//   Bt: f16, PRE-TRANSPOSED [n][K] so every tile copy is contiguous b128
//   C : f32, row stride ldc
// 256 threads = 8 waves; each wave owns a 16x64 strip = 4 tiles of
// v_wmma_f32_16x16x32_f16, f32 accumulation. K-step 32.
// ------------------------------------------------------------------
template<int NCOLS, int WAVES_M, int WAVES_N, int ACT, int AF16>
__global__ __launch_bounds__(256) void gemm_wmma(
    const void* __restrict__ Av, int lda,
    const _Float16* __restrict__ Bt,
    const float* __restrict__ bias,
    float* __restrict__ C, int ldc, int K)
{
    constexpr int MT = WAVES_M * 16;
    __shared__ _Float16 Alds[MT * 32];       // [row][k]
    __shared__ _Float16 Blds[NCOLS * 32];    // [col][k]

    const int tid  = threadIdx.x;
    const int wave = tid >> 5, lane = tid & 31;
    const int wm   = wave / WAVES_N, wn = wave % WAVES_N;
    const int half = lane >> 4, l16 = lane & 15;
    const long rowblk = (long)blockIdx.x * MT;

    v8f acc[4] = {};

    for (int k0 = 0; k0 < K; k0 += 32) {
        __syncthreads();
        if (AF16) {
            // A already f16: contiguous 16B chunks -> async LDS DMA
            const _Float16* A = (const _Float16*)Av;
            #pragma unroll
            for (int i = tid * 8; i < MT * 32; i += 256 * 8) {
                int r = i >> 5, kk = i & 31;
                copy16_g2lds(&A[(rowblk + r) * (long)lda + k0 + kk], &Alds[i]);
            }
        } else {
            // A f32: float4 load, pack to 4xf16, one b64 LDS store
            const float* A = (const float*)Av;
            #pragma unroll
            for (int i = tid * 4; i < MT * 32; i += 256 * 4) {
                int r = i >> 5, kk = i & 31;
                const float* src = &A[(rowblk + r) * (long)lda + k0 + kk];
                if (k0 + 32 < K) __builtin_prefetch(src + 32, 0, 0);
                float4 f = *(const float4*)src;
                H4 p;
                p.h[0] = (_Float16)f.x; p.h[1] = (_Float16)f.y;
                p.h[2] = (_Float16)f.z; p.h[3] = (_Float16)f.w;
                *(uint2*)&Alds[i] = p.u2;
            }
        }
        // B tile: pre-transposed f16, contiguous 16B chunks
        #pragma unroll
        for (int i = tid * 8; i < NCOLS * 32; i += 256 * 8) {
            int n = i >> 5, kk = i & 31;
            copy16_g2lds(&Bt[(long)n * K + k0 + kk], &Blds[i]);
        }
        wait_async_copies();
        __syncthreads();

        // A fragment: ISA 16-bit layout (VGPR v -> K pair 2v / 16+2(v-4), +8*half)
        HFrag af;
        const int arow = wm * 16 + l16;
        #pragma unroll
        for (int v = 0; v < 8; ++v) {
            int kk = (v < 4 ? 2 * v : 16 + 2 * (v - 4)) + 8 * half;
            af.u[v] = *(const unsigned*)&Alds[arow * 32 + kk];
        }
        #pragma unroll
        for (int t = 0; t < 4; ++t) {
            HFrag bf;
            const int ncol = wn * 64 + t * 16 + l16;
            #pragma unroll
            for (int v = 0; v < 8; ++v) {
                int kk = (v < 4 ? 2 * v : 16 + 2 * (v - 4)) + 8 * half;
                bf.u[v] = *(const unsigned*)&Blds[ncol * 32 + kk];
            }
            acc[t] = __builtin_amdgcn_wmma_f32_16x16x32_f16(
                false, af.v, false, bf.v, (short)0, acc[t], false, false);
        }
    }

    // epilogue: C layout (VGPR r, lane L) -> row r + 8*half, col L%16
    #pragma unroll
    for (int t = 0; t < 4; ++t) {
        const int col = wn * 64 + t * 16 + l16;
        const float bv = bias ? bias[col] : 0.f;
        #pragma unroll
        for (int r = 0; r < 8; ++r) {
            long row = rowblk + wm * 16 + r + 8 * half;
            float val = acc[t][r] + bv;
            if (ACT == ACT_RELU)    val = fmaxf(val, 0.f);
            if (ACT == ACT_SIGMOID) val = 1.f / (1.f + __expf(-val));
            C[row * (long)ldc + col] = val;
        }
    }
}

// Bt[n*K + k] = (f16) B[k*N + n]   -- one-time weight transpose+convert
__global__ __launch_bounds__(256) void k_trans(
    const float* __restrict__ B, _Float16* __restrict__ Bt, int K, int N)
{
    int i = blockIdx.x * 256 + threadIdx.x;
    if (i >= K * N) return;
    int k = i / N, n = i % N;
    Bt[(long)n * K + k] = (_Float16)B[i];
}

// f16(relu(x)) conversion, float4-vectorized (fuses the GAT layer ReLU)
__global__ __launch_bounds__(256) void k_cvt_relu(
    const float* __restrict__ in, _Float16* __restrict__ outh)
{
    long i = ((long)blockIdx.x * 256 + threadIdx.x) * 4;
    float4 f = *(const float4*)&in[i];
    H4 p;
    p.h[0] = (_Float16)fmaxf(f.x, 0.f); p.h[1] = (_Float16)fmaxf(f.y, 0.f);
    p.h[2] = (_Float16)fmaxf(f.z, 0.f); p.h[3] = (_Float16)fmaxf(f.w, 0.f);
    *(uint2*)&outh[i] = p.u2;
}

// ------------------- order-preserving float<->uint ----------------
__device__ __forceinline__ unsigned fenc(float f) {
    unsigned u = __float_as_uint(f);
    return (u & 0x80000000u) ? ~u : (u ^ 0x80000000u);
}
__device__ __forceinline__ float fdec(unsigned u) {
    unsigned b = (u & 0x80000000u) ? (u ^ 0x80000000u) : ~u;
    return __uint_as_float(b);
}

// es = h @ a_src, ed = h @ a_dst (wave per row, shuffle reduce)
__global__ __launch_bounds__(256) void k_dots(
    const float* __restrict__ h, const float* __restrict__ a_s,
    const float* __restrict__ a_d, float* __restrict__ es, float* __restrict__ ed)
{
    int row  = (blockIdx.x * 256 + threadIdx.x) >> 5;
    int lane = threadIdx.x & 31;
    const float* p = h + (long)row * HC;
    float ss = 0.f, sd = 0.f;
    #pragma unroll
    for (int k = lane; k < HC; k += 32) { float x = p[k]; ss += x * a_s[k]; sd += x * a_d[k]; }
    #pragma unroll
    for (int o = 16; o > 0; o >>= 1) { ss += __shfl_down(ss, o); sd += __shfl_down(sd, o); }
    if (lane == 0) { es[row] = ss; ed[row] = sd; }
}

// out[n, c] = bias[c]   (seed for the scatter-add aggregation)
__global__ __launch_bounds__(256) void k_init_bias(float* __restrict__ out, const float* __restrict__ b)
{
    long i = ((long)blockIdx.x * 256 + threadIdx.x) * 4;
    *(float4*)&out[i] = *(const float4*)&b[(int)(i & (HC - 1))];
}

__global__ void k_init_sm(unsigned* __restrict__ menc, float* __restrict__ den)
{
    int i = blockIdx.x * 256 + threadIdx.x;
    menc[i] = 0u;           // < fenc(x) for every finite x => acts as -inf
    den[i]  = 0.f;
}

// per-edge raw score + segment max over dst
__global__ __launch_bounds__(256) void k_score(
    const int* __restrict__ ei, const float* __restrict__ es,
    const float* __restrict__ ed, float* __restrict__ sc, unsigned* __restrict__ menc)
{
    int e = blockIdx.x * 256 + threadIdx.x;
    int s, d;
    if (e < NEDGES) { s = ei[e]; d = ei[NEDGES + e]; } else { s = d = e - NEDGES; }
    float v = es[s] + ed[d];
    v = v > 0.f ? v : 0.2f * v;          // leaky_relu(0.2)
    sc[e] = v;
    atomicMax(&menc[d], fenc(v));
}

// exp(score - max[dst]) + segment sum over dst
__global__ __launch_bounds__(256) void k_exp(
    const int* __restrict__ ei, float* __restrict__ sc,
    const unsigned* __restrict__ menc, float* __restrict__ den)
{
    int e = blockIdx.x * 256 + threadIdx.x;
    int d = (e < NEDGES) ? ei[NEDGES + e] : (e - NEDGES);
    float ex = __expf(sc[e] - fdec(menc[d]));
    sc[e] = ex;
    atomicAdd(&den[d], ex);
}

// out[dst] += alpha * h[src] -- 32 lanes/edge, float4 gather, 4 atomics/lane
__global__ __launch_bounds__(256) void k_agg(
    const int* __restrict__ ei, const float* __restrict__ h,
    const float* __restrict__ sc, const float* __restrict__ den,
    float* __restrict__ out)
{
    long tid = (long)blockIdx.x * 256 + threadIdx.x;
    long e  = tid >> 5;
    int  c4 = (int)(tid & 31) * 4;
    int s, d;
    if (e < NEDGES) { s = ei[e]; d = ei[NEDGES + e]; } else { s = d = (int)(e - NEDGES); }
    float alpha = sc[e] / (den[d] + 1e-16f);
    float4 hv = *(const float4*)&h[(long)s * HC + c4];
    float* o = out + (long)d * HC + c4;
    atomicAdd(o + 0, alpha * hv.x);
    atomicAdd(o + 1, alpha * hv.y);
    atomicAdd(o + 2, alpha * hv.z);
    atomicAdd(o + 3, alpha * hv.w);
}

// global max pool, ReLU folded in (max(relu(x)) == relu(max(x)))
__global__ __launch_bounds__(128) void k_pool(const float* __restrict__ h, float* __restrict__ pooled)
{
    int g = blockIdx.x, c = threadIdx.x;
    const float* p = h + (long)g * NPG * HC + c;
    float m = -INFINITY;
    for (int r = 0; r < NPG; ++r) m = fmaxf(m, p[(long)r * HC]);
    pooled[g * HC + c] = fmaxf(m, 0.f);
}

// fused[i, col_off + j] = sum_t tanh(eh[i,t] + ee[j,t]) * v[t]   (ba folded into ee)
__global__ __launch_bounds__(32) void k_bahdanau(
    const float* __restrict__ eh, const float* __restrict__ ee,
    const float* __restrict__ v, float* __restrict__ out, int ld_out, int col_off)
{
    int i = blockIdx.x, lane = threadIdx.x;
    float e0 = eh[i * OUTC + lane], e1 = eh[i * OUTC + lane + 32];
    float v0 = v[lane], v1 = v[lane + 32];
    for (int j = 0; j < NG; ++j) {
        float p = tanhf(e0 + ee[j * OUTC + lane]) * v0
                + tanhf(e1 + ee[j * OUTC + lane + 32]) * v1;
        #pragma unroll
        for (int o = 16; o > 0; o >>= 1) p += __shfl_down(p, o);
        if (lane == 0) out[i * ld_out + col_off + j] = p;
    }
}

// ==================================================================
extern "C" void kernel_launch(void* const* d_in, const int* in_sizes, int n_in,
                              void* d_out, int out_size, void* d_ws, size_t ws_size,
                              hipStream_t stream)
{
    (void)in_sizes; (void)n_in; (void)out_size; (void)ws_size;
    const float* x   = (const float*)d_in[0];
    const int*   ei  = (const int*)  d_in[1];
    const float* W1  = (const float*)d_in[3];
    const float* as1 = (const float*)d_in[4];
    const float* ad1 = (const float*)d_in[5];
    const float* b1  = (const float*)d_in[6];
    const float* W2  = (const float*)d_in[7];
    const float* as2 = (const float*)d_in[8];
    const float* ad2 = (const float*)d_in[9];
    const float* b2  = (const float*)d_in[10];
    const float* W3  = (const float*)d_in[11];
    const float* as3 = (const float*)d_in[12];
    const float* ad3 = (const float*)d_in[13];
    const float* b3  = (const float*)d_in[14];
    const float* Wn  = (const float*)d_in[15];
    const float* bn  = (const float*)d_in[16];
    const float* W0  = (const float*)d_in[17];
    const float* b0  = (const float*)d_in[18];
    const float* Wl  = (const float*)d_in[19];
    const float* bl  = (const float*)d_in[20];
    const float* Wa1 = (const float*)d_in[21];
    const float* ba1 = (const float*)d_in[22];
    const float* v1  = (const float*)d_in[23];
    const float* Wa2 = (const float*)d_in[24];
    const float* ba2 = (const float*)d_in[25];
    const float* v2  = (const float*)d_in[26];
    float* outp = (float*)d_out;

    // ----- workspace layout (~176 MB) -----
    char* w = (char*)d_ws;
    float*    hA     = (float*)w;    w += (size_t)NNODES * HC * 4;   // GEMM out (pre-agg h)
    float*    hB     = (float*)w;    w += (size_t)NNODES * HC * 4;   // aggregation out
    _Float16* hBh    = (_Float16*)w; w += (size_t)NNODES * HC * 2;   // relu(hB) as f16
    float*    es     = (float*)w;    w += (size_t)NNODES * 4;
    float*    ed     = (float*)w;    w += (size_t)NNODES * 4;
    unsigned* menc   = (unsigned*)w; w += (size_t)NNODES * 4;
    float*    den    = (float*)w;    w += (size_t)NNODES * 4;
    float*    sc     = (float*)w;    w += (size_t)ETOT * 4;
    float*    pooled = (float*)w;    w += (size_t)NG * HC * 4;
    float*    hidden = (float*)w;    w += (size_t)NG * HC * 4;
    float*    news   = (float*)w;    w += (size_t)NG * HC * 4;
    float*    eh     = (float*)w;    w += (size_t)NG * OUTC * 4;
    float*    ee     = (float*)w;    w += (size_t)NG * OUTC * 4;
    float*    fused  = (float*)w;    w += (size_t)NG * 2 * HC * 4;
    _Float16* W1t    = (_Float16*)w; w += (size_t)INC * HC * 2;
    _Float16* W2t    = (_Float16*)w; w += (size_t)HC * HC * 2;
    _Float16* W3t    = (_Float16*)w; w += (size_t)HC * HC * 2;
    _Float16* Wnt    = (_Float16*)w; w += (size_t)INC * HC * 2;
    _Float16* W0t    = (_Float16*)w; w += (size_t)HC * HC * 2;
    _Float16* Wa1ta  = (_Float16*)w; w += (size_t)HC * OUTC * 2;
    _Float16* Wa1tb  = (_Float16*)w; w += (size_t)HC * OUTC * 2;
    _Float16* Wa2ta  = (_Float16*)w; w += (size_t)HC * OUTC * 2;
    _Float16* Wa2tb  = (_Float16*)w; w += (size_t)HC * OUTC * 2;
    _Float16* Wlt    = (_Float16*)w; w += (size_t)(2 * HC) * OUTC * 2;

    // ----- one-time weight transpose+convert to f16 [n][K] -----
    #define TR(B, Bt, K, N) k_trans<<<((K)*(N)+255)/256,256,0,stream>>>(B, Bt, K, N);
    TR(W1, W1t, INC, HC)         TR(W2, W2t, HC, HC)   TR(W3, W3t, HC, HC)
    TR(Wn, Wnt, INC, HC)         TR(W0, W0t, HC, HC)
    TR(Wa1, Wa1ta, HC, OUTC)     TR(Wa1 + HC * OUTC, Wa1tb, HC, OUTC)
    TR(Wa2, Wa2ta, HC, OUTC)     TR(Wa2 + HC * OUTC, Wa2tb, HC, OUTC)
    TR(Wl, Wlt, 2 * HC, OUTC)
    #undef TR

    const int gNH4 = (NNODES * HC) / 1024;   // float4 elementwise over N*H
    const int gN   = NNODES / 256;
    const int gE   = ETOT / 256;
    const int gAgg = (int)(((long)ETOT * 32) / 256);

    // ---------------- GAT layers ----------------
    // layer 1: A = x (f32, K=256); layers 2/3: A = hBh (f16, K=128)
    #define GAT_TAIL(AS, AD, Bv)                                              \
        k_dots<<<NNODES/8,256,0,stream>>>(hA, AS, AD, es, ed);                \
        k_init_bias<<<gNH4,256,0,stream>>>(hB, Bv);                           \
        k_init_sm<<<gN,256,0,stream>>>(menc, den);                            \
        k_score<<<gE,256,0,stream>>>(ei, es, ed, sc, menc);                   \
        k_exp<<<gE,256,0,stream>>>(ei, sc, menc, den);                        \
        k_agg<<<gAgg,256,0,stream>>>(ei, hA, sc, den, hB);

    gemm_wmma<128,4,2,ACT_NONE,0><<<NNODES/64,256,0,stream>>>(x, INC, W1t, nullptr, hA, HC, INC);
    GAT_TAIL(as1, ad1, b1)
    k_cvt_relu<<<gNH4,256,0,stream>>>(hB, hBh);

    gemm_wmma<128,4,2,ACT_NONE,1><<<NNODES/64,256,0,stream>>>(hBh, HC, W2t, nullptr, hA, HC, HC);
    GAT_TAIL(as2, ad2, b2)
    k_cvt_relu<<<gNH4,256,0,stream>>>(hB, hBh);

    gemm_wmma<128,4,2,ACT_NONE,1><<<NNODES/64,256,0,stream>>>(hBh, HC, W3t, nullptr, hA, HC, HC);
    GAT_TAIL(as3, ad3, b3)
    #undef GAT_TAIL

    // global max pool (ReLU of layer 3 folded in) -> (G,H)
    k_pool<<<NG,128,0,stream>>>(hB, pooled);

    // hidden = relu(pooled @ W0 + b0)
    gemm_wmma<128,4,2,ACT_RELU,0><<<NG/64,256,0,stream>>>(pooled, HC, W0t, b0, hidden, HC, HC);

    // news = relu(x[root] @ Wn + bn); root g = g*NPG -> stride-lda row gather
    gemm_wmma<128,4,2,ACT_RELU,0><<<NG/64,256,0,stream>>>(x, NPG*INC, Wnt, bn, news, HC, INC);

    // fused_h = bahdanau(hidden, news, Wa1, ba1, v1) -> fused[:, 0:128]
    gemm_wmma<64,8,1,ACT_NONE,0><<<1,256,0,stream>>>(hidden, HC, Wa1ta, nullptr, eh, OUTC, HC);
    gemm_wmma<64,8,1,ACT_NONE,0><<<1,256,0,stream>>>(news,   HC, Wa1tb, ba1,     ee, OUTC, HC);
    k_bahdanau<<<NG,32,0,stream>>>(eh, ee, v1, fused, 2*HC, 0);

    // fused_news = bahdanau(news, hidden, Wa2, ba2, v2) -> fused[:, 128:256]
    gemm_wmma<64,8,1,ACT_NONE,0><<<1,256,0,stream>>>(news,   HC, Wa2ta, nullptr, eh, OUTC, HC);
    gemm_wmma<64,8,1,ACT_NONE,0><<<1,256,0,stream>>>(hidden, HC, Wa2tb, ba2,     ee, OUTC, HC);
    k_bahdanau<<<NG,32,0,stream>>>(eh, ee, v2, fused, 2*HC, HC);

    // out = sigmoid(fused @ Wl + bl)
    gemm_wmma<64,8,1,ACT_SIGMOID,0><<<1,256,0,stream>>>(fused, 2*HC, Wlt, bl, outp, OUTC, 2*HC);
}
